// PretrainedCROMA_7473243095470
// MI455X (gfx1250) — matrix operand order repaired and checked by reading, not compiled, not run on hardware
//
#include <hip/hip_runtime.h>
#include <hip/hip_bf16.h>
#include <math.h>

// ---------------------------------------------------------------------------
// Types for CDNA5 WMMA (wave32) + TDM descriptors
// ---------------------------------------------------------------------------
typedef __attribute__((ext_vector_type(16))) __bf16 v16bf;
typedef __attribute__((ext_vector_type(8)))  __bf16 v8bf;
typedef __attribute__((ext_vector_type(8)))  float  v8f;
typedef __attribute__((ext_vector_type(4)))  unsigned int v4u;
typedef __attribute__((ext_vector_type(8)))  int v8i_t;
typedef __attribute__((ext_vector_type(4)))  int v4i_t;

#if __has_builtin(__builtin_amdgcn_tensor_load_to_lds) && __has_builtin(__builtin_amdgcn_s_wait_tensorcnt)
#define CROMA_TDM 1
#else
#define CROMA_TDM 0
#endif

// ---------------------------------------------------------------------------
// Model constants
// ---------------------------------------------------------------------------
#define DIMV    768
#define HEADS_  16
#define NP      225          // patches per image
#define BQ      8            // batch
#define ROWS_   (BQ * NP)    // 1800
#define DH      48           // head dim
#define NPAD    240          // padded 225 -> keeps bf16 rows 16B aligned
#define BH      (BQ * HEADS_) // 128 batched heads
#define DFF     3072

// ---------------------------------------------------------------------------
// TDM: async 2D bf16 tile copy global->LDS.
//   tile: tile_m rows x 32 cols (bf16), row stride lda elements in memory.
//   LDS layout: rows of 64B + 16B pad -> 80B stride (matches [..][40] arrays).
//   OOB (beyond Mrem/Krem) reads return zero -> hardware edge zero-fill.
//   This toolchain declares the 6-arg builtin:
//     (uint32x4 g0, int32x8 g1, int32x4 g2, int32x4 g3, int32x8 g4, i32 cpol)
//   Groups 2/3/4 are zero (2D tensor, no gather/iterate).
// ---------------------------------------------------------------------------
#if CROMA_TDM
__device__ __forceinline__ void tdm_load_tile(unsigned lds_off, const __bf16* gptr,
                                              int Krem, int Mrem, int lda_elems,
                                              int tile_m)
{
    const unsigned long long ga = (unsigned long long)(uintptr_t)gptr;
    v4u g0;
    g0[0] = 1u;                                        // count=1 (valid user D#)
    g0[1] = lds_off;                                   // lds_addr (bytes)
    g0[2] = (unsigned)(ga & 0xFFFFFFFFu);              // global_addr[31:0]
    g0[3] = (unsigned)((ga >> 32) & 0x1FFFFFFu)        // global_addr[56:32]
          | (2u << 30);                                // type = 2 ("image")
    v8i_t g1;
    // data_size=1 (2B) | pad_enable | pad_interval=3 (16 DW) | pad_amount=3 (4 DW)
    g1[0] = (int)((1u << 16) | (1u << 20) | (3u << 22) | (3u << 25));
    g1[1] = (int)(((unsigned)Krem & 0xFFFFu) << 16);                 // tensor_dim0[15:0]
    g1[2] = (int)((((unsigned)Krem >> 16) & 0xFFFFu)                 // tensor_dim0[31:16]
          | (((unsigned)Mrem & 0xFFFFu) << 16));                     // tensor_dim1[15:0]
    g1[3] = (int)((((unsigned)Mrem >> 16) & 0xFFFFu)                 // tensor_dim1[31:16]
          | (32u << 16));                                            // tile_dim0 = 32
    g1[4] = (int)((unsigned)tile_m & 0xFFFFu);                       // tile_dim1 (tile_dim2=0)
    g1[5] = lda_elems;                                               // tensor_dim0_stride lo32
    g1[6] = 0;                                                       // stride hi / dim1_stride lo
    g1[7] = 0;
    const v4i_t z4 = {0, 0, 0, 0};
    const v8i_t z8 = {0, 0, 0, 0, 0, 0, 0, 0};
    __builtin_amdgcn_tensor_load_to_lds(g0, g1, z4, z4, z8, 0);
}
#endif

// ---------------------------------------------------------------------------
// Generic tiled WMMA GEMM:
//   C[M,N] = epilogue( A[M,K](bf16,row-major) x B )
// BMODE==0 : B given as Bt[N,K] bf16 row-major (already "transposed")
// BMODE==1 : B given as  B[K,N] f32 row-major  (weights; transpose+cvt in LDS)
// Tile 128x64x32, 256 threads = 8 waves, each wave: 16(M) x 64(N), 4 WMMAs/step
// A tile (and Bt tile in BMODE==0) staged by TDM async DMA; f32 weights
// transposed+converted by VALU while the TDM copy is in flight.
// ---------------------------------------------------------------------------
template<int BMODE, bool HAS_BIAS, bool HAS_BMAT, bool HAS_RES, bool HAS_GELU, bool OUT_BF16>
__global__ __launch_bounds__(256)
void gemm_wmma(const __bf16* __restrict__ A, long sA, int lda,
               const void* __restrict__ Bv, long sB, int ldb,
               void* __restrict__ Cv, int cDiv, long cSO, long cSI, int ldc,
               const float* __restrict__ bias,
               const float* __restrict__ bmat, int bmMod, long bmStride, int bmLd,
               float scale, int M, int N, int K)
{
    __shared__ __bf16 As[128][40];   // 32 + 8 pad halves -> 80B row stride
    __shared__ __bf16 Bs[64][40];

    const int t  = threadIdx.x;
    const int z  = blockIdx.z;
    const int m0 = blockIdx.x * 128;
    const int n0 = blockIdx.y * 64;

    A += (long)z * sA;
    const __bf16* Bt = (const __bf16*)Bv + ((BMODE == 0) ? (long)z * sB : 0);
    const float*  Bf = (const float*) Bv + ((BMODE == 1) ? (long)z * sB : 0);

    const long coff = (long)(z / cDiv) * cSO + (long)(z % cDiv) * cSI;
    float*  Cf = (float*) Cv + coff;
    __bf16* Cb = (__bf16*)Cv + coff;
    const float* bm = HAS_BMAT ? (bmat + (long)(z % bmMod) * bmStride) : nullptr;

    const int wid = t >> 5;        // wave id 0..7
    const int lane = t & 31;
    const int lh = lane >> 4;      // 0/1 half of wave
    const int lm = lane & 15;

    v8f acc[4];
#pragma unroll
    for (int i = 0; i < 4; ++i) {
#pragma unroll
        for (int j = 0; j < 8; ++j) acc[i][j] = 0.f;
    }

#if CROMA_TDM
    const unsigned as_off = (unsigned)(uintptr_t)&As[0][0];
    const unsigned bs_off = (unsigned)(uintptr_t)&Bs[0][0];
#endif

    for (int k0 = 0; k0 < K; k0 += 32) {
#if CROMA_TDM
        // ---- issue async TDM tile copies (wave 0 only; EXEC-independent DMA) ----
        if (wid == 0) {
            tdm_load_tile(as_off, A + (long)m0 * lda + k0, K - k0, M - m0, lda, 128);
            if (BMODE == 0)
                tdm_load_tile(bs_off, Bt + (long)n0 * ldb + k0, K - k0, N - n0, ldb, 64);
        }
#else
        // ---- manual A tile: 128 x 32 bf16, each thread moves 16 halves ----
        {
            const int m  = t >> 1;
            const int kc = (t & 1) * 16;
            const int gm = m0 + m;
            const __bf16* src = A + (long)gm * lda + k0 + kc;
            if (gm < M && (k0 + kc + 16) <= K) {
                *(v8bf*)&As[m][kc]     = *(const v8bf*)(src);
                *(v8bf*)&As[m][kc + 8] = *(const v8bf*)(src + 8);
            } else {
#pragma unroll
                for (int j = 0; j < 16; ++j)
                    As[m][kc + j] = (gm < M && (k0 + kc + j) < K) ? src[j] : (__bf16)0.f;
            }
        }
        if (BMODE == 0) {
            const int n  = t >> 2;
            const int kc = (t & 3) * 8;
            const int gn = n0 + n;
            const __bf16* src = Bt + (long)gn * ldb + k0 + kc;
            if (gn < N && (k0 + kc + 8) <= K) {
                *(v8bf*)&Bs[n][kc] = *(const v8bf*)src;
            } else {
#pragma unroll
                for (int j = 0; j < 8; ++j)
                    Bs[n][kc + j] = (gn < N && (k0 + kc + j) < K) ? src[j] : (__bf16)0.f;
            }
        }
#endif
        // ---- f32 weights: transpose + convert into Bs (overlaps TDM) ----
        if (BMODE == 1) {
            const int k  = t >> 3;
            const int nc = (t & 7) * 8;
            const int gk = k0 + k;
            const float* src = Bf + (long)gk * ldb + n0 + nc;
#pragma unroll
            for (int j = 0; j < 8; ++j) {
                float v = (gk < K && (n0 + nc + j) < N) ? src[j] : 0.f;
                Bs[nc + j][k] = (__bf16)v;
            }
            // prefetch next weight tile (global_prefetch_b8)
            if (k0 + 32 + k < K) __builtin_prefetch(Bf + (long)(k0 + 32 + k) * ldb + n0, 0, 1);
        }
#if CROMA_TDM
        if (wid == 0) __builtin_amdgcn_s_wait_tensorcnt(0);  // LDS tiles landed
#endif
        __syncthreads();

        // ---- A fragment: lane<16 -> K{0..7,16..23}; lane>=16 -> K{8..15,24..31}
        union { v16bf v; v8bf h[2]; } af;
        {
            const int row = wid * 16 + lm;
            af.h[0] = *(const v8bf*)&As[row][lh * 8];
            af.h[1] = *(const v8bf*)&As[row][16 + lh * 8];
        }
        // ---- B fragments + WMMA ----
#pragma unroll
        for (int nt = 0; nt < 4; ++nt) {
            union { v16bf v; v8bf h[2]; } bf_;
            const int col = nt * 16 + lm;
            bf_.h[0] = *(const v8bf*)&Bs[col][lh * 8];
            bf_.h[1] = *(const v8bf*)&Bs[col][16 + lh * 8];
            acc[nt] = __builtin_amdgcn_wmma_f32_16x16x32_bf16(
                false, af.v, false, bf_.v, (short)0, acc[nt], false, false);
        }
        __syncthreads();
    }

    // ---- epilogue: lane=N, vgpr r = M row (lanes>=16: +8) ----
#pragma unroll
    for (int nt = 0; nt < 4; ++nt) {
        const int gn = n0 + nt * 16 + lm;
        if (gn >= N) continue;
        const float bi = HAS_BIAS ? bias[gn] : 0.f;
#pragma unroll
        for (int r = 0; r < 8; ++r) {
            const int gm = m0 + wid * 16 + lh * 8 + r;
            if (gm >= M) continue;
            float v = acc[nt][r];
            if (HAS_BMAT) v = v * scale + bm[(long)gm * bmLd + gn];
            if (HAS_BIAS) v += bi;
            if (HAS_RES)  v += Cf[(long)gm * ldc + gn];
            if (HAS_GELU) v = 0.5f * v * (1.f + erff(v * 0.70710678118654752f));
            if (OUT_BF16) Cb[(long)gm * ldc + gn] = (__bf16)v;
            else          Cf[(long)gm * ldc + gn] = v;
        }
    }
}

// ---------------------------------------------------------------------------
// LayerNorm (row-wise over D), f32 in -> bf16 or f32 out
// ---------------------------------------------------------------------------
template<bool OBF>
__global__ __launch_bounds__(256)
void ln_kernel(const float* __restrict__ x, const float* __restrict__ g,
               const float* __restrict__ b, void* __restrict__ out, int D, int ld)
{
    const int row = blockIdx.x;
    const float* xr = x + (long)row * D;
    __shared__ float s1[256], s2[256];
    float a0 = 0.f, a1 = 0.f;
    for (int i = threadIdx.x; i < D; i += 256) { float v = xr[i]; a0 += v; a1 += v * v; }
    s1[threadIdx.x] = a0; s2[threadIdx.x] = a1;
    __syncthreads();
    for (int st = 128; st > 0; st >>= 1) {
        if (threadIdx.x < st) { s1[threadIdx.x] += s1[threadIdx.x + st]; s2[threadIdx.x] += s2[threadIdx.x + st]; }
        __syncthreads();
    }
    const float mean = s1[0] / D;
    const float var  = s2[0] / D - mean * mean;
    const float rstd = rsqrtf(var + 1e-5f);
    for (int i = threadIdx.x; i < D; i += 256) {
        const float y = (xr[i] - mean) * rstd * g[i] + b[i];
        if (OBF) ((__bf16*)out)[(long)row * ld + i] = (__bf16)y;
        else     ((float*) out)[(long)row * ld + i] = y;
    }
}

// ---------------------------------------------------------------------------
// Softmax over 225 cols, f32 in -> bf16 out (row stride ldp)
// ---------------------------------------------------------------------------
__global__ __launch_bounds__(256)
void softmax_kernel(const float* __restrict__ s, __bf16* __restrict__ p,
                    int cols, int lds, int ldp)
{
    const int row = blockIdx.x;
    const float* sr = s + (long)row * lds;
    __shared__ float red[256];
    const int i = threadIdx.x;
    const float v = (i < cols) ? sr[i] : -3.4e38f;
    red[i] = v; __syncthreads();
    for (int st = 128; st > 0; st >>= 1) { if (i < st) red[i] = fmaxf(red[i], red[i + st]); __syncthreads(); }
    const float m = red[0]; __syncthreads();
    const float e = (i < cols) ? __expf(v - m) : 0.f;
    red[i] = e; __syncthreads();
    for (int st = 128; st > 0; st >>= 1) { if (i < st) red[i] += red[i + st]; __syncthreads(); }
    const float inv = 1.f / red[0];
    if (i < cols) p[(long)row * ldp + i] = (__bf16)(e * inv);
}

// ---------------------------------------------------------------------------
// mean over patch axis: [Bq,Nn,D] -> [Bq,D]
// ---------------------------------------------------------------------------
__global__ void mean_kernel(const float* __restrict__ x, float* __restrict__ out,
                            int Bq, int Nn, int D)
{
    const int idx = blockIdx.x * blockDim.x + threadIdx.x;
    if (idx >= Bq * D) return;
    const int b = idx / D, d = idx % D;
    const float* p = x + (long)b * Nn * D + d;
    float s = 0.f;
    for (int n = 0; n < Nn; ++n) s += p[(long)n * D];
    out[idx] = s / (float)Nn;
}

// ---------------------------------------------------------------------------
// patchify: imgs [B,C,120,120] -> bf16 [B*225, C*64] with (c,i,j) layout
// ---------------------------------------------------------------------------
__global__ void patchify_kernel(const float* __restrict__ img, __bf16* __restrict__ out, int C)
{
    const long total = (long)BQ * NP * C * 64;
    const long idx = (long)blockIdx.x * blockDim.x + threadIdx.x;
    if (idx >= total) return;
    const int  j = (int)(idx & 7);
    long r = idx >> 3;
    const int  i = (int)(r & 7); r >>= 3;
    const int  c = (int)(r % C); r /= C;
    const int  p = (int)(r % NP);
    const int  b = (int)(r / NP);
    const int hh = p / 15, ww = p % 15;
    out[idx] = (__bf16)img[(((long)b * C + c) * 120 + hh * 8 + i) * 120 + ww * 8 + j];
}

// ---------------------------------------------------------------------------
// head-split kernels (bf16)
// ---------------------------------------------------------------------------
__global__ void split_qkv_kernel(const __bf16* __restrict__ qkv, __bf16* __restrict__ q,
                                 __bf16* __restrict__ k, __bf16* __restrict__ vT)
{
    const long total = (long)ROWS_ * 2304;
    const long idx = (long)blockIdx.x * blockDim.x + threadIdx.x;
    if (idx >= total) return;
    const int col = (int)(idx % 2304);
    const long row = idx / 2304;
    const int b = (int)(row / NP), nn = (int)(row % NP);
    const int part = col / DIMV, rem = col % DIMV;
    const int h = rem / DH, d = rem % DH;
    const __bf16 v = qkv[idx];
    const long bh = (long)b * HEADS_ + h;
    if (part == 0)      q[(bh * NP + nn) * DH + d] = v;
    else if (part == 1) k[(bh * NP + nn) * DH + d] = v;
    else                vT[(bh * DH + d) * NPAD + nn] = v;
}

__global__ void split_heads_kernel(const __bf16* __restrict__ in, __bf16* __restrict__ out)
{
    const long total = (long)ROWS_ * DIMV;
    const long idx = (long)blockIdx.x * blockDim.x + threadIdx.x;
    if (idx >= total) return;
    const int col = (int)(idx % DIMV);
    const long row = idx / DIMV;
    const int b = (int)(row / NP), nn = (int)(row % NP);
    const int h = col / DH, d = col % DH;
    out[(((long)b * HEADS_ + h) * NP + nn) * DH + d] = in[idx];
}

__global__ void split_vT_kernel(const __bf16* __restrict__ in, __bf16* __restrict__ vT)
{
    const long total = (long)ROWS_ * DIMV;
    const long idx = (long)blockIdx.x * blockDim.x + threadIdx.x;
    if (idx >= total) return;
    const int col = (int)(idx % DIMV);
    const long row = idx / DIMV;
    const int b = (int)(row / NP), nn = (int)(row % NP);
    const int h = col / DH, d = col % DH;
    vT[(((long)b * HEADS_ + h) * DH + d) * NPAD + nn] = in[idx];
}

// ---------------------------------------------------------------------------
// Host-side parameter structs and orchestration
// ---------------------------------------------------------------------------
struct AttnP  { const float *ln_g, *ln_b, *qkv_w, *out_w, *out_b; };
struct XAttnP { const float *ln_g, *ln_b, *q_w, *k_w, *v_w, *out_w, *out_b; };
struct FfnP   { const float *ln_g, *ln_b, *fc1_w, *fc1_b, *fc2_w, *fc2_b; };
struct GapP   { const float *ln_g, *ln_b, *fc1_w, *fc1_b, *fc2_w, *fc2_b; };
struct VitP   { const float *in_w, *in_b; AttnP attn[12]; FfnP ffn[12]; const float *ln_g, *ln_b; int depth; };
struct CrossP { AttnP sa[6]; XAttnP ca[6]; FfnP ffn[6]; const float *ln_g, *ln_b; };

struct WS {
    __bf16 *xn, *cn, *qkv, *qd, *kd, *vd, *q, *k, *vT, *probs, *o, *hbuf, *pbuf, *gln, *ghid;
    float  *scores, *gmean, *x1, *x2, *xj;
};

struct Binder {
    void* const* in; int cur;
    const float* nx() { return (const float*)in[cur++]; }
    void attn(AttnP& a)  { a.ln_g=nx(); a.ln_b=nx(); a.qkv_w=nx(); a.out_w=nx(); a.out_b=nx(); }
    void xattn(XAttnP& a){ a.ln_g=nx(); a.ln_b=nx(); a.q_w=nx(); a.k_w=nx(); a.v_w=nx(); a.out_w=nx(); a.out_b=nx(); }
    void ffn(FfnP& f)    { f.ln_g=nx(); f.ln_b=nx(); f.fc1_w=nx(); f.fc1_b=nx(); f.fc2_w=nx(); f.fc2_b=nx(); }
    void gap(GapP& g)    { g.ln_g=nx(); g.ln_b=nx(); g.fc1_w=nx(); g.fc1_b=nx(); g.fc2_w=nx(); g.fc2_b=nx(); }
    void vit(VitP& v, int depth) {
        v.depth = depth; v.in_w = nx(); v.in_b = nx();
        for (int i = 0; i < depth; ++i) { attn(v.attn[i]); ffn(v.ffn[i]); }
        v.ln_g = nx(); v.ln_b = nx();
    }
    void cross(CrossP& c) {
        for (int i = 0; i < 6; ++i) { attn(c.sa[i]); xattn(c.ca[i]); ffn(c.ffn[i]); }
        c.ln_g = nx(); c.ln_b = nx();
    }
};

template<int BMODE, bool BIAS, bool BMAT, bool RES, bool GELU, bool OBF>
static inline void gemm(hipStream_t st, const __bf16* A, long sA, int lda,
                        const void* B, long sB, int ldb,
                        void* C, int cDiv, long cSO, long cSI, int ldc,
                        const float* bias, const float* bmat, int bmMod, long bmStride, int bmLd,
                        float scale, int M, int N, int K, int Z)
{
    dim3 g((M + 127) / 128, (N + 63) / 64, Z);
    gemm_wmma<BMODE, BIAS, BMAT, RES, GELU, OBF><<<g, 256, 0, st>>>(
        A, sA, lda, B, sB, ldb, C, cDiv, cSO, cSI, ldc,
        bias, bmat, bmMod, bmStride, bmLd, scale, M, N, K);
}

static const float ATTN_SCALE = 0.14433756729740645f; // 48^-0.5

static void attn_core(hipStream_t st, float* x, const float* out_w, const float* out_b,
                      const float* alibi, WS& w)
{
    // scores: q @ k^T * scale + alibi[h]   (batched over 128 b*h)
    gemm<0, false, true, false, false, false>(st, w.q, (long)NP * DH, DH,
        w.k, (long)NP * DH, DH, w.scores, 1, (long)NP * NP, 0, NP,
        nullptr, alibi, HEADS_, (long)NP * NP, NP, ATTN_SCALE, NP, NP, DH, BH);
    softmax_kernel<<<BH * NP, 256, 0, st>>>(w.scores, w.probs, NP, NP, NPAD);
    // o[b, m, h*48+n] = probs @ V
    gemm<0, false, false, false, false, true>(st, w.probs, (long)NP * NPAD, NPAD,
        w.vT, (long)DH * NPAD, NPAD, w.o, HEADS_, (long)NP * DIMV, DH, DIMV,
        nullptr, nullptr, 1, 0, 0, 0.f, NP, DH, NP, BH);
    // x += o @ Wout + b
    gemm<1, true, false, true, false, false>(st, w.o, 0, DIMV, out_w, 0, DIMV,
        x, 1, 0, 0, DIMV, out_b, nullptr, 1, 0, 0, 0.f, ROWS_, DIMV, DIMV, 1);
}

static void attn_block(hipStream_t st, float* x, const AttnP& p, const float* alibi, WS& w)
{
    ln_kernel<true><<<ROWS_, 256, 0, st>>>(x, p.ln_g, p.ln_b, w.xn, DIMV, DIMV);
    gemm<1, false, false, false, false, true>(st, w.xn, 0, DIMV, p.qkv_w, 0, 3 * DIMV,
        w.qkv, 1, 0, 0, 3 * DIMV, nullptr, nullptr, 1, 0, 0, 0.f, ROWS_, 3 * DIMV, DIMV, 1);
    const long tot = (long)ROWS_ * 2304;
    split_qkv_kernel<<<(int)((tot + 255) / 256), 256, 0, st>>>(w.qkv, w.q, w.k, w.vT);
    attn_core(st, x, p.out_w, p.out_b, alibi, w);
}

static void xattn_block(hipStream_t st, float* x, const float* ctx, const XAttnP& p,
                        const float* alibi, WS& w)
{
    ln_kernel<true><<<ROWS_, 256, 0, st>>>(x,   p.ln_g, p.ln_b, w.xn, DIMV, DIMV);
    ln_kernel<true><<<ROWS_, 256, 0, st>>>(ctx, p.ln_g, p.ln_b, w.cn, DIMV, DIMV);
    gemm<1, false, false, false, false, true>(st, w.xn, 0, DIMV, p.q_w, 0, DIMV,
        w.qd, 1, 0, 0, DIMV, nullptr, nullptr, 1, 0, 0, 0.f, ROWS_, DIMV, DIMV, 1);
    gemm<1, false, false, false, false, true>(st, w.cn, 0, DIMV, p.k_w, 0, DIMV,
        w.kd, 1, 0, 0, DIMV, nullptr, nullptr, 1, 0, 0, 0.f, ROWS_, DIMV, DIMV, 1);
    gemm<1, false, false, false, false, true>(st, w.cn, 0, DIMV, p.v_w, 0, DIMV,
        w.vd, 1, 0, 0, DIMV, nullptr, nullptr, 1, 0, 0, 0.f, ROWS_, DIMV, DIMV, 1);
    const long tot = (long)ROWS_ * DIMV;
    const int nb = (int)((tot + 255) / 256);
    split_heads_kernel<<<nb, 256, 0, st>>>(w.qd, w.q);
    split_heads_kernel<<<nb, 256, 0, st>>>(w.kd, w.k);
    split_vT_kernel  <<<nb, 256, 0, st>>>(w.vd, w.vT);
    attn_core(st, x, p.out_w, p.out_b, alibi, w);
}

static void ffn_block(hipStream_t st, float* x, const FfnP& p, WS& w)
{
    ln_kernel<true><<<ROWS_, 256, 0, st>>>(x, p.ln_g, p.ln_b, w.xn, DIMV, DIMV);
    gemm<1, true, false, false, true, true>(st, w.xn, 0, DIMV, p.fc1_w, 0, DFF,
        w.hbuf, 1, 0, 0, DFF, p.fc1_b, nullptr, 1, 0, 0, 0.f, ROWS_, DFF, DIMV, 1);
    gemm<1, true, false, true, false, false>(st, w.hbuf, 0, DFF, p.fc2_w, 0, DIMV,
        x, 1, 0, 0, DIMV, p.fc2_b, nullptr, 1, 0, 0, 0.f, ROWS_, DIMV, DFF, 1);
}

static void vit_forward(hipStream_t st, const float* imgs, int C, const VitP& p,
                        const float* alibi, float* x, float* enc_out, WS& w)
{
    const long tot = (long)ROWS_ * C * 64;
    patchify_kernel<<<(int)((tot + 255) / 256), 256, 0, st>>>(imgs, w.pbuf, C);
    gemm<1, true, false, false, false, false>(st, w.pbuf, 0, C * 64, p.in_w, 0, DIMV,
        x, 1, 0, 0, DIMV, p.in_b, nullptr, 1, 0, 0, 0.f, ROWS_, DIMV, C * 64, 1);
    for (int d = 0; d < p.depth; ++d) {
        attn_block(st, x, p.attn[d], alibi, w);
        ffn_block (st, x, p.ffn[d],  w);
    }
    ln_kernel<false><<<ROWS_, 256, 0, st>>>(x, p.ln_g, p.ln_b, enc_out, DIMV, DIMV);
}

static void gap_forward(hipStream_t st, const float* enc, const GapP& p, float* out, WS& w)
{
    mean_kernel<<<(BQ * DIMV + 255) / 256, 256, 0, st>>>(enc, w.gmean, BQ, NP, DIMV);
    ln_kernel<true><<<BQ, 256, 0, st>>>(w.gmean, p.ln_g, p.ln_b, w.gln, DIMV, DIMV);
    gemm<1, true, false, false, true, true>(st, w.gln, 0, DIMV, p.fc1_w, 0, DFF,
        w.ghid, 1, 0, 0, DFF, p.fc1_b, nullptr, 1, 0, 0, 0.f, BQ, DFF, DIMV, 1);
    gemm<1, true, false, false, false, false>(st, w.ghid, 0, DFF, p.fc2_w, 0, DIMV,
        out, 1, 0, 0, DIMV, p.fc2_b, nullptr, 1, 0, 0, 0.f, BQ, DIMV, DFF, 1);
}

// ---------------------------------------------------------------------------
// entry point
// ---------------------------------------------------------------------------
extern "C" void kernel_launch(void* const* d_in, const int* in_sizes, int n_in,
                              void* d_out, int out_size, void* d_ws, size_t ws_size,
                              hipStream_t stream)
{
    (void)in_sizes; (void)n_in; (void)out_size; (void)ws_size;

    const float* sar_imgs = (const float*)d_in[0];
    const float* opt_imgs = (const float*)d_in[1];
    const float* alibi    = (const float*)d_in[2];

    Binder bd{d_in, 3};
    static VitP s1, s2; static CrossP cr; static GapP g1, g2;
    bd.vit(s1, 6); bd.vit(s2, 12); bd.cross(cr); bd.gap(g1); bd.gap(g2);

    // workspace bump allocator (256B aligned)
    char* base = (char*)d_ws; size_t off = 0;
    auto alloc = [&](size_t bytes) -> void* {
        void* p = base + off; off += (bytes + 255) & ~(size_t)255; return p;
    };
    WS w;
    w.xn     = (__bf16*)alloc((size_t)ROWS_ * DIMV * 2);
    w.cn     = (__bf16*)alloc((size_t)ROWS_ * DIMV * 2);
    w.qkv    = (__bf16*)alloc((size_t)ROWS_ * 2304 * 2);
    w.qd     = (__bf16*)alloc((size_t)ROWS_ * DIMV * 2);
    w.kd     = (__bf16*)alloc((size_t)ROWS_ * DIMV * 2);
    w.vd     = (__bf16*)alloc((size_t)ROWS_ * DIMV * 2);
    w.q      = (__bf16*)alloc((size_t)BH * NP * DH * 2);
    w.k      = (__bf16*)alloc((size_t)BH * NP * DH * 2);
    w.vT     = (__bf16*)alloc((size_t)BH * DH * NPAD * 2);
    w.probs  = (__bf16*)alloc((size_t)BH * NP * NPAD * 2);
    w.o      = (__bf16*)alloc((size_t)ROWS_ * DIMV * 2);
    w.hbuf   = (__bf16*)alloc((size_t)ROWS_ * DFF * 2);
    w.pbuf   = (__bf16*)alloc((size_t)ROWS_ * DIMV * 2);
    w.gln    = (__bf16*)alloc((size_t)BQ * DIMV * 2);
    w.ghid   = (__bf16*)alloc((size_t)BQ * DFF * 2);
    w.scores = (float*)alloc((size_t)BH * NP * NP * 4);
    w.gmean  = (float*)alloc((size_t)BQ * DIMV * 4);
    w.x1     = (float*)alloc((size_t)ROWS_ * DIMV * 4);
    w.x2     = (float*)alloc((size_t)ROWS_ * DIMV * 4);
    w.xj     = (float*)alloc((size_t)ROWS_ * DIMV * 4);

    // output regions (all f32, concatenated in return order)
    float* out_f = (float*)d_out;
    float* sar_enc  = out_f;                              // 8*225*768
    float* sar_gap  = sar_enc + (size_t)ROWS_ * DIMV;     // 8*768
    float* opt_enc  = sar_gap + (size_t)BQ * DIMV;
    float* opt_gap  = opt_enc + (size_t)ROWS_ * DIMV;
    float* joint    = opt_gap + (size_t)BQ * DIMV;
    float* joint_gap= joint   + (size_t)ROWS_ * DIMV;

    // SAR ViT + GAP
    vit_forward(stream, sar_imgs, 2, s1, alibi, w.x1, sar_enc, w);
    gap_forward(stream, sar_enc, g1, sar_gap, w);
    // OPT ViT + GAP
    vit_forward(stream, opt_imgs, 12, s2, alibi, w.x2, opt_enc, w);
    gap_forward(stream, opt_enc, g2, opt_gap, w);

    // Cross encoder: x starts at sar_enc, ctx = opt_enc
    (void)hipMemcpyAsync(w.xj, sar_enc, (size_t)ROWS_ * DIMV * 4, hipMemcpyDeviceToDevice, stream);
    for (int d = 0; d < 6; ++d) {
        attn_block (stream, w.xj, cr.sa[d],  alibi, w);
        xattn_block(stream, w.xj, opt_enc, cr.ca[d], alibi, w);
        ffn_block  (stream, w.xj, cr.ffn[d], w);
    }
    ln_kernel<false><<<ROWS_, 256, 0, stream>>>(w.xj, cr.ln_g, cr.ln_b, joint, DIMV, DIMV);
    mean_kernel<<<(BQ * DIMV + 255) / 256, 256, 0, stream>>>(joint, joint_gap, BQ, NP, DIMV);
}
